// TransformerDecoder_67078799229233
// MI455X (gfx1250) — matrix operand order repaired and checked by reading, not compiled, not run on hardware
//
#include <hip/hip_runtime.h>

// ---------------------------------------------------------------------------
// Transformer decoder for MI455X (gfx1250).
// All heavy math goes through v_wmma_f32_16x16x32_bf16 (wave32 WMMA).
// fp32 tensors are converted to bf16 (v_cvt_pk_bf16_f32) while staging into
// LDS; LDS tiles are laid out so every WMMA fragment load is ds_load_b128.
// The GEMM K-loop is software-pipelined with double-buffered LDS so global
// loads for tile t+1 overlap the WMMA work on tile t.
// ---------------------------------------------------------------------------

typedef __attribute__((ext_vector_type(16))) __bf16 v16bf;
typedef __attribute__((ext_vector_type(8)))  __bf16 v8bf;
typedef __attribute__((ext_vector_type(4)))  __bf16 v4bf;
typedef __attribute__((ext_vector_type(8)))  float  v8f;

// Model constants (match reference)
#define BB   4
#define SS   512
#define DD   1024
#define HH   16
#define HDD  64
#define LL   6
#define VV   32000
#define FFD  4096
#define ROWS (BB * SS)   // 2048 token rows

// ---- fp32 -> bf16 via native packed converts ------------------------------
__device__ __forceinline__ void cvt_store4(__bf16* p, float4 f) {
  v4bf v;
  v[0] = (__bf16)f.x; v[1] = (__bf16)f.y;
  v[2] = (__bf16)f.z; v[3] = (__bf16)f.w;
  *reinterpret_cast<v4bf*>(p) = v;      // 2x v_cvt_pk_bf16_f32 + ds_store_b64
}

// ---------------------------------------------------------------------------
// Tiled WMMA GEMM:  C[m,n] = act( alpha * sum_k A[m,k]*B[k,n] + bias[n] )
//   - 256 threads = 8 wave32s in a 4(row) x 2(col) grid
//   - block tile BM x BN, K step 32; wave tile (BM/4) x (BN/2)
//   - A tile in LDS: As[m][k]   (row-major, 16B-aligned runs per fragment)
//   - B tile in LDS: Bt[n][k]   (k-contiguous per column -> b128 frag loads)
//   - double-buffered LDS + reg-staged global loads: loads for tile t+1
//     issue before the barrier/compute of tile t (one barrier per k-step)
//   - TRANS_B: B element (k,n) read from Bw[n*ldb + k]   (used for Q.K^T)
//   - CAUSAL : write -1e20 where key(n) > query(m)       (self-attention)
//   - batch via blockIdx.z decomposed as (b = z/Hdiv, h = z%Hdiv)
// ---------------------------------------------------------------------------
template<int BM, int BN, bool TRANS_B, bool RELU, bool CAUSAL>
__global__ __launch_bounds__(256) void gemm_k(
    const float* __restrict__ Ab, int lda, long aSB, long aSH,
    const float* __restrict__ Bb, int ldb, long bSB, long bSH,
    const float* __restrict__ bias,
    float* __restrict__ Cb, int ldc, long cSB, long cSH,
    int K, float alpha, int Hdiv)
{
  constexpr int BK = 32;
  constexpr int LDT = BK + 8;            // 40 elems = 80B rows (16B multiple)
  constexpr int WROWS = 4, WCOLS = 2;
  constexpr int WTM = BM / WROWS, WTN = BN / WCOLS;
  constexpr int MFR = WTM / 16,  NFR = WTN / 16;
  constexpr int ANUM = (BM * BK / 4) / 256;
  constexpr int BNUM = (BN * BK / 4) / 256;
  constexpr int BNV  = BN / 4;

  __shared__ __align__(16) __bf16 As[2][BM][LDT];
  __shared__ __align__(16) __bf16 Bt[2][BN][LDT];

  const int z  = blockIdx.z;
  const int bb = z / Hdiv, hh = z - bb * Hdiv;
  const float* A  = Ab + (long)bb * aSB + (long)hh * aSH;
  const float* Bw = Bb + (long)bb * bSB + (long)hh * bSH;
  float*       C  = Cb + (long)bb * cSB + (long)hh * cSH;

  const int m0  = blockIdx.y * BM;
  const int n0  = blockIdx.x * BN;
  const int tid = threadIdx.x;
  const int wid = tid >> 5, lane = tid & 31;   // wave32
  const int wr  = wid & 3,  wc   = wid >> 2;
  const int lf  = lane & 15, lh  = lane >> 4;

  // ---- thread-fixed staging addresses (computed once) ----------------------
  const float* aptr[ANUM]; __bf16* asl[ANUM];
#pragma unroll
  for (int t = 0; t < ANUM; ++t) {
    const int i = tid + t * 256, m = i >> 3, k4 = (i & 7) << 2;
    aptr[t] = &A[(long)(m0 + m) * lda + k4];
    asl[t]  = &As[0][m][k4];
  }
  const float* bptr[BNUM]; __bf16* bsl[BNUM];
  if constexpr (TRANS_B) {
#pragma unroll
    for (int t = 0; t < BNUM; ++t) {
      const int i = tid + t * 256, n = i >> 3, k4 = (i & 7) << 2;
      bptr[t] = &Bw[(long)(n0 + n) * ldb + k4];
      bsl[t]  = &Bt[0][n][k4];
    }
  } else {
#pragma unroll
    for (int t = 0; t < BNUM; ++t) {
      const int i = tid + t * 256, k = i / BNV, n4 = (i - k * BNV) << 2;
      bptr[t] = &Bw[(long)k * ldb + n0 + n4];
      bsl[t]  = &Bt[0][n4][k];
    }
  }
  // thread-fixed fragment read addresses
  const __bf16* arow[MFR]; const __bf16* bcol[NFR];
#pragma unroll
  for (int mi = 0; mi < MFR; ++mi) arow[mi] = &As[0][wr * WTM + mi * 16 + lf][0];
#pragma unroll
  for (int ni = 0; ni < NFR; ++ni) bcol[ni] = &Bt[0][wc * WTN + ni * 16 + lf][0];

  float4 ra[ANUM], rb[BNUM];

  // batched global loads for tile starting at kt (+ prefetch one tile further)
  auto loadG = [&](int kt) {
    const long ob = TRANS_B ? (long)kt : (long)kt * ldb;
#pragma unroll
    for (int t = 0; t < ANUM; ++t)
      ra[t] = *reinterpret_cast<const float4*>(aptr[t] + kt);
#pragma unroll
    for (int t = 0; t < BNUM; ++t)
      rb[t] = *reinterpret_cast<const float4*>(bptr[t] + ob);
    if (kt + BK < K) {                   // uniform guard, one branch per step
      const long ob2 = TRANS_B ? (long)(kt + BK) : (long)(kt + BK) * ldb;
#pragma unroll
      for (int t = 0; t < ANUM; ++t) __builtin_prefetch(aptr[t] + kt + BK, 0, 3);
#pragma unroll
      for (int t = 0; t < BNUM; ++t) __builtin_prefetch(bptr[t] + ob2, 0, 3);
    }
  };
  // regs -> LDS (bf16) into buffer `buf`
  auto storeL = [&](int buf) {
    const int ao = buf * (BM * LDT), bo = buf * (BN * LDT);
#pragma unroll
    for (int t = 0; t < ANUM; ++t) cvt_store4(asl[t] + ao, ra[t]);
    if constexpr (TRANS_B) {
#pragma unroll
      for (int t = 0; t < BNUM; ++t) cvt_store4(bsl[t] + bo, rb[t]);
    } else {
#pragma unroll
      for (int t = 0; t < BNUM; ++t) {
        __bf16* p = bsl[t] + bo;
        p[0 * LDT] = (__bf16)rb[t].x;
        p[1 * LDT] = (__bf16)rb[t].y;
        p[2 * LDT] = (__bf16)rb[t].z;
        p[3 * LDT] = (__bf16)rb[t].w;
      }
    }
  };

  v8f acc[MFR][NFR];
#pragma unroll
  for (int i = 0; i < MFR; ++i)
#pragma unroll
    for (int j = 0; j < NFR; ++j) acc[i][j] = v8f{};

  // ---- software-pipelined K loop ------------------------------------------
  loadG(0);
  storeL(0);
  int cur = 0;
  for (int kt = 0; kt < K; kt += BK) {
    const bool more = (kt + BK) < K;
    if (more) loadG(kt + BK);            // issue next tile's global loads
    __syncthreads();                     // buffer `cur` ready for everyone

    // fragment loads (all ds_load_b128) + 8 WMMAs on buffer `cur`
    // A 16x32 bf16: lane half h -> runs K=[8h,8h+7] and K=[16+8h,16+8h+7]
    // B 32x16 bf16: lane half h -> runs K=[8h,8h+7] and K=[16+8h,16+8h+7], column = lane%16
    const int ao = cur * (BM * LDT), bo = cur * (BN * LDT);
    v16bf afr[MFR], bfr[NFR];
#pragma unroll
    for (int mi = 0; mi < MFR; ++mi) {
      const __bf16* row = arow[mi] + ao;
      const v8bf lo = *reinterpret_cast<const v8bf*>(row + (lh << 3));
      const v8bf hi = *reinterpret_cast<const v8bf*>(row + 16 + (lh << 3));
      afr[mi] = __builtin_shufflevector(lo, hi, 0, 1, 2, 3, 4, 5, 6, 7,
                                        8, 9, 10, 11, 12, 13, 14, 15);
    }
#pragma unroll
    for (int ni = 0; ni < NFR; ++ni) {
      const __bf16* col = bcol[ni] + bo;
      const v8bf lo = *reinterpret_cast<const v8bf*>(col + (lh << 3));
      const v8bf hi = *reinterpret_cast<const v8bf*>(col + 16 + (lh << 3));
      bfr[ni] = __builtin_shufflevector(lo, hi, 0, 1, 2, 3, 4, 5, 6, 7,
                                        8, 9, 10, 11, 12, 13, 14, 15);
    }
#pragma unroll
    for (int mi = 0; mi < MFR; ++mi)
#pragma unroll
      for (int ni = 0; ni < NFR; ++ni)
        acc[mi][ni] = __builtin_amdgcn_wmma_f32_16x16x32_bf16(
            false, afr[mi], false, bfr[ni], (short)0, acc[mi][ni],
            false, false);

    if (more) storeL(cur ^ 1);           // waits on this tile's global loads
    cur ^= 1;
  }

  // ---- epilogue: C 16x16 f32 layout (VGPR r -> M = r + 8h, lane%16 -> N) ---
#pragma unroll
  for (int mi = 0; mi < MFR; ++mi) {
#pragma unroll
    for (int ni = 0; ni < NFR; ++ni) {
      const int gn = n0 + wc * WTN + ni * 16 + lf;
      const float bv = bias ? bias[gn] : 0.0f;
#pragma unroll
      for (int r = 0; r < 8; ++r) {
        const int gm = m0 + wr * WTM + mi * 16 + (lh << 3) + r;
        float v = acc[mi][ni][r] * alpha + bv;
        if constexpr (RELU)   v = fmaxf(v, 0.0f);
        if constexpr (CAUSAL) { if (gn > gm) v = -1e20f; }
        C[(long)gm * ldc + gn] = v;
      }
    }
  }
}

// ---------------------------------------------------------------------------
// Row softmax, row length 512. One wave32 per row, 8 rows per block.
// ---------------------------------------------------------------------------
__global__ __launch_bounds__(256) void softmax512(float* __restrict__ E) {
  const int wid  = threadIdx.x >> 5, lane = threadIdx.x & 31;
  const long row = (long)blockIdx.x * 8 + wid;
  float* p = E + row * 512;
  float v[16];
  float mx = -3.0e38f;
#pragma unroll
  for (int i = 0; i < 16; ++i) { v[i] = p[lane + (i << 5)]; mx = fmaxf(mx, v[i]); }
#pragma unroll
  for (int o = 16; o; o >>= 1) mx = fmaxf(mx, __shfl_xor(mx, o, 32));
  float s = 0.0f;
#pragma unroll
  for (int i = 0; i < 16; ++i) { v[i] = __expf(v[i] - mx); s += v[i]; }
#pragma unroll
  for (int o = 16; o; o >>= 1) s += __shfl_xor(s, o, 32);
  const float inv = 1.0f / s;
#pragma unroll
  for (int i = 0; i < 16; ++i) p[lane + (i << 5)] = v[i] * inv;
}

// ---------------------------------------------------------------------------
// out = LayerNorm(a + b) * g + beta     (one 1024-wide row per block)
// Safe for out aliasing a or b: all reads precede writes via the barrier.
// ---------------------------------------------------------------------------
__global__ __launch_bounds__(256) void add_ln(
    const float* __restrict__ a, const float* __restrict__ b,
    const float* __restrict__ g, const float* __restrict__ be,
    float* __restrict__ o) {
  const long row = blockIdx.x;
  const float* pa = a + row * DD;
  const float* pb = b + row * DD;
  float* po = o + row * DD;
  float x[4];
  float s = 0.0f, s2 = 0.0f;
#pragma unroll
  for (int j = 0; j < 4; ++j) {
    const int c = threadIdx.x + (j << 8);
    x[j] = pa[c] + pb[c];
    s += x[j]; s2 += x[j] * x[j];
  }
  const int wid = threadIdx.x >> 5, lane = threadIdx.x & 31;
#pragma unroll
  for (int o2 = 16; o2; o2 >>= 1) {
    s  += __shfl_xor(s,  o2, 32);
    s2 += __shfl_xor(s2, o2, 32);
  }
  __shared__ float rs[8], rq[8];
  if (lane == 0) { rs[wid] = s; rq[wid] = s2; }
  __syncthreads();
  float ts = 0.0f, tq = 0.0f;
#pragma unroll
  for (int w = 0; w < 8; ++w) { ts += rs[w]; tq += rq[w]; }
  const float mean = ts * (1.0f / DD);
  const float var  = tq * (1.0f / DD) - mean * mean;
  const float rstd = rsqrtf(var + 1e-5f);
#pragma unroll
  for (int j = 0; j < 4; ++j) {
    const int c = threadIdx.x + (j << 8);
    po[c] = (x[j] - mean) * rstd * g[c] + be[c];
  }
}

// ---------------------------------------------------------------------------
// x = emb[trg] + pos[s]        (one 1024-wide row per block)
// ---------------------------------------------------------------------------
__global__ __launch_bounds__(256) void embed_k(
    const int* __restrict__ trg, const float* __restrict__ emb,
    const float* __restrict__ pos, float* __restrict__ X) {
  const long row = blockIdx.x;                 // b*S + s
  const long s   = row & (SS - 1);
  const long tok = trg[row];
#pragma unroll
  for (int j = 0; j < 4; ++j) {
    const int c = threadIdx.x + (j << 8);
    X[row * DD + c] = emb[tok * DD + c] + pos[s * DD + c];
  }
}

// ---------------------------------------------------------------------------
// Host orchestration
// ---------------------------------------------------------------------------
static void launch_gemm(const float* A, const float* W, const float* bias,
                        float* C, int Mrows, int N, int K, bool relu,
                        hipStream_t stream) {
  dim3 grid(N / 128, Mrows / 128, 1);
  if (relu)
    gemm_k<128, 128, false, true, false><<<grid, 256, 0, stream>>>(
        A, K, 0, 0, W, N, 0, 0, bias, C, N, 0, 0, K, 1.0f, 1);
  else
    gemm_k<128, 128, false, false, false><<<grid, 256, 0, stream>>>(
        A, K, 0, 0, W, N, 0, 0, bias, C, N, 0, 0, K, 1.0f, 1);
}

static void launch_qkt(const float* Q, const float* Kp, float* E, bool causal,
                       hipStream_t stream) {
  dim3 grid(SS / 128, SS / 128, BB * HH);
  const float scale = 0.125f;                    // 1/sqrt(64)
  if (causal)
    gemm_k<128, 128, true, false, true><<<grid, 256, 0, stream>>>(
        Q,  DD, (long)SS * DD, HDD,
        Kp, DD, (long)SS * DD, HDD,
        nullptr,
        E,  SS, (long)HH * SS * SS, (long)SS * SS,
        HDD, scale, HH);
  else
    gemm_k<128, 128, true, false, false><<<grid, 256, 0, stream>>>(
        Q,  DD, (long)SS * DD, HDD,
        Kp, DD, (long)SS * DD, HDD,
        nullptr,
        E,  SS, (long)HH * SS * SS, (long)SS * SS,
        HDD, scale, HH);
}

static void launch_av(const float* E, const float* Vp, float* AO,
                      hipStream_t stream) {
  dim3 grid(HDD / 64, SS / 128, BB * HH);
  gemm_k<128, 64, false, false, false><<<grid, 256, 0, stream>>>(
      E,  SS, (long)HH * SS * SS, (long)SS * SS,
      Vp, DD, (long)SS * DD, HDD,
      nullptr,
      AO, DD, (long)SS * DD, HDD,
      SS, 1.0f, HH);
}

extern "C" void kernel_launch(void* const* d_in, const int* in_sizes, int n_in,
                              void* d_out, int out_size, void* d_ws, size_t ws_size,
                              hipStream_t stream) {
  (void)in_sizes; (void)n_in; (void)out_size; (void)ws_size;

  const int*   trg    = (const int*)  d_in[0];
  const float* memory = (const float*)d_in[1];
  // d_in[2] = trg_mask: pure causal tril -> applied analytically (key > query)
  const float* emb    = (const float*)d_in[3];
  const float* pos    = (const float*)d_in[4];
  const float* out_w  = (const float*)d_in[5];
  const float* out_b  = (const float*)d_in[6];
  const float* sa_wq = (const float*)d_in[7];  const float* sa_bq = (const float*)d_in[8];
  const float* sa_wk = (const float*)d_in[9];  const float* sa_bk = (const float*)d_in[10];
  const float* sa_wv = (const float*)d_in[11]; const float* sa_bv = (const float*)d_in[12];
  const float* sa_wo = (const float*)d_in[13]; const float* sa_bo = (const float*)d_in[14];
  const float* ca_wq = (const float*)d_in[15]; const float* ca_bq = (const float*)d_in[16];
  const float* ca_wk = (const float*)d_in[17]; const float* ca_bk = (const float*)d_in[18];
  const float* ca_wv = (const float*)d_in[19]; const float* ca_bv = (const float*)d_in[20];
  const float* ca_wo = (const float*)d_in[21]; const float* ca_bo = (const float*)d_in[22];
  const float* ln1_g = (const float*)d_in[23]; const float* ln1_b = (const float*)d_in[24];
  const float* ln2_g = (const float*)d_in[25]; const float* ln2_b = (const float*)d_in[26];
  const float* ln3_g = (const float*)d_in[27]; const float* ln3_b = (const float*)d_in[28];
  const float* ff_w1 = (const float*)d_in[29]; const float* ff_b1 = (const float*)d_in[30];
  const float* ff_w2 = (const float*)d_in[31]; const float* ff_b2 = (const float*)d_in[32];

  // Workspace layout (fp32)
  float* ws = (float*)d_ws;
  float* X  = ws;                               // [2048,1024] current x
  float* U  = X  + (long)ROWS * DD;             // [2048,1024] post-ln1 "q"
  float* Q  = U  + (long)ROWS * DD;             // [2048,1024]
  float* Kp = Q  + (long)ROWS * DD;             // [2048,1024]
  float* Vp = Kp + (long)ROWS * DD;             // [2048,1024]
  float* AO = Vp + (long)ROWS * DD;             // [2048,1024] attn concat out
  float* T  = AO + (long)ROWS * DD;             // [2048,1024] temp
  float* F1 = T  + (long)ROWS * DD;             // [2048,4096] ffn hidden
  float* E  = F1 + (long)ROWS * FFD;            // [4,16,512,512] scores

  embed_k<<<ROWS, 256, 0, stream>>>(trg, emb, pos, X);

  const int nSMrows = (BB * HH * SS) / 8;       // 4096 blocks, 8 rows each

  for (int i = 0; i < LL; ++i) {
    const long wOff = (long)i * DD * DD;
    const long bOff = (long)i * DD;

    // ---- self attention (causal) ----
    launch_gemm(X, sa_wq + wOff, sa_bq + bOff, Q,  ROWS, DD, DD, false, stream);
    launch_gemm(X, sa_wk + wOff, sa_bk + bOff, Kp, ROWS, DD, DD, false, stream);
    launch_gemm(X, sa_wv + wOff, sa_bv + bOff, Vp, ROWS, DD, DD, false, stream);
    launch_qkt(Q, Kp, E, /*causal=*/true, stream);
    softmax512<<<nSMrows, 256, 0, stream>>>(E);
    launch_av(E, Vp, AO, stream);
    launch_gemm(AO, sa_wo + wOff, sa_bo + bOff, T, ROWS, DD, DD, false, stream);
    add_ln<<<ROWS, 256, 0, stream>>>(T, X, ln1_g + bOff, ln1_b + bOff, U);

    // ---- cross attention (memory K/V, no mask) ----
    launch_gemm(U,      ca_wq + wOff, ca_bq + bOff, Q,  ROWS, DD, DD, false, stream);
    launch_gemm(memory, ca_wk + wOff, ca_bk + bOff, Kp, ROWS, DD, DD, false, stream);
    launch_gemm(memory, ca_wv + wOff, ca_bv + bOff, Vp, ROWS, DD, DD, false, stream);
    launch_qkt(Q, Kp, E, /*causal=*/false, stream);
    softmax512<<<nSMrows, 256, 0, stream>>>(E);
    launch_av(E, Vp, AO, stream);
    launch_gemm(AO, ca_wo + wOff, ca_bo + bOff, T, ROWS, DD, DD, false, stream);
    add_ln<<<ROWS, 256, 0, stream>>>(T, U, ln3_g + bOff, ln3_b + bOff, X);

    // ---- feed forward ----
    launch_gemm(X,  ff_w1 + (long)i * DD * FFD, ff_b1 + (long)i * FFD,
                F1, ROWS, FFD, DD, /*relu=*/true, stream);
    launch_gemm(F1, ff_w2 + (long)i * FFD * DD, ff_b2 + bOff,
                T,  ROWS, DD, FFD, false, stream);
    add_ln<<<ROWS, 256, 0, stream>>>(T, X, ln2_g + bOff, ln2_b + bOff, X);
  }

  // ---- final logits: [2048,1024] @ [1024,32000] + out_b -> d_out ----
  launch_gemm(X, out_w, out_b, (float*)d_out, ROWS, VV, DD, false, stream);
}
